// GraphTransformer_52999896433289
// MI455X (gfx1250) — compile-verified, matrix operand order checked by reading, not run on hardware
//
#include <hip/hip_runtime.h>
#include <hip/hip_bf16.h>

// ---------------------------------------------------------------------------
// GraphTransformer (3x TransformerConv + BN/ReLU) for MI455X / gfx1250.
//   * Dense projections: register-blocked bf16 WMMA GEMM, one wave owns a
//     16-row x NC output stripe (NC/16 accumulators live across K), B staged
//     transposed+bf16 in LDS (aligned ds_load_b128 fragment reads), fused
//     Q/K/V/skip variant amortizes A reads 4x.
//   * Attention: encoded-uint atomicMax segment max, f32 atomicAdd scatters.
//   * BatchNorm: per-channel LDS tree reduction.
// ---------------------------------------------------------------------------

typedef __attribute__((ext_vector_type(16))) __bf16 v16bf;
typedef __attribute__((ext_vector_type(8)))  __bf16 v8bf;
typedef __attribute__((ext_vector_type(8)))  float  v8f;

__device__ __forceinline__ v16bf load_a_frag(const float* __restrict__ ar) {
  // ar points at A[row, 32*f + khalf]; lane needs K offsets {0..7} and {16..23}.
  float4 f0 = *(const float4*)(ar);
  float4 f1 = *(const float4*)(ar + 4);
  float4 f2 = *(const float4*)(ar + 16);
  float4 f3 = *(const float4*)(ar + 20);
  v16bf a;
  a[0]  = (__bf16)f0.x; a[1]  = (__bf16)f0.y; a[2]  = (__bf16)f0.z; a[3]  = (__bf16)f0.w;
  a[4]  = (__bf16)f1.x; a[5]  = (__bf16)f1.y; a[6]  = (__bf16)f1.z; a[7]  = (__bf16)f1.w;
  a[8]  = (__bf16)f2.x; a[9]  = (__bf16)f2.y; a[10] = (__bf16)f2.z; a[11] = (__bf16)f2.w;
  a[12] = (__bf16)f3.x; a[13] = (__bf16)f3.y; a[14] = (__bf16)f3.z; a[15] = (__bf16)f3.w;
  return a;
}

// ---------------------------------------------------------------------------
// C[o][M,NC] = A[M,K] * W[o][K,NC] + bias[o]  for o < NOUT (NOUT=4 fuses the
// Q/K/V/skip projections over shared A). Requires M%16==0; K%32==0; NC in
// {32,256}. 256 threads = 8 waves; wave w handles row tile blockIdx.x*8+w
// (clamped so EXEC/barrier membership stay full) and the entire NC width.
// ---------------------------------------------------------------------------
template <int K, int NC, int NOUT>
__global__ __launch_bounds__(256)
void gemm_wmma(const float* __restrict__ A, int M,
               const float* W0, const float* W1, const float* W2, const float* W3,
               const float* B0, const float* B1, const float* B2, const float* B3,
               float* C0, float* C1, float* C2, float* C3) {
  constexpr int NK = K / 32;        // K-steps
  constexpr int NT = NC / 16;       // column tiles per wave
  constexpr int LSTRIDE = 40;       // bf16 elements per staged column (80B pad)
  __shared__ __bf16 sw[NC * LSTRIDE];

  const int lane  = threadIdx.x & 31;
  const int wave  = threadIdx.x >> 5;
  const int tilesM = M >> 4;
  int r = blockIdx.x * 8 + wave;
  if (r >= tilesM) r = tilesM - 1;  // clamp: duplicate tile, identical stores

  const int l16   = lane & 15;
  const int khalf = (lane >> 4) << 3;          // 0 or 8 (ISA 16-bit layout)
  const int row   = r * 16 + l16;

  const float* Ws_[4] = {W0, W1, W2, W3};
  const float* Bs_[4] = {B0, B1, B2, B3};
  float*       Cs_[4] = {C0, C1, C2, C3};

#pragma unroll
  for (int o = 0; o < NOUT; ++o) {
    const float* __restrict__ W = Ws_[o];
    v8f acc[NT];
#pragma unroll
    for (int t = 0; t < NT; ++t) acc[t] = (v8f){};

    for (int f = 0; f < NK; ++f) {
      __syncthreads();              // protect previous slice's readers
      // Cooperative stage: 32 x NC slice of W, transposed + bf16, coalesced.
      for (int i = threadIdx.x; i < 32 * NC; i += 256) {
        const int k = i / NC;
        const int c = i % NC;       // consecutive threads -> consecutive c
        sw[c * LSTRIDE + k] = (__bf16)W[(size_t)(f * 32 + k) * NC + c];
      }
      __syncthreads();

      const v16bf a = load_a_frag(A + (size_t)row * K + f * 32 + khalf);
#pragma unroll
      for (int t = 0; t < NT; ++t) {
        const __bf16* cb = sw + (t * 16 + l16) * LSTRIDE + khalf;
        v8bf lo = *(const v8bf*)(cb);        // ds_load_b128 (16B aligned)
        v8bf hi = *(const v8bf*)(cb + 16);   // ds_load_b128
        v16bf b = __builtin_shufflevector(lo, hi, 0, 1, 2, 3, 4, 5, 6, 7,
                                          8, 9, 10, 11, 12, 13, 14, 15);
        acc[t] = __builtin_amdgcn_wmma_f32_16x16x32_bf16(
            false, a, false, b, (short)0, acc[t], false, false);
      }
    }

    float* __restrict__ C = Cs_[o];
    const float* bias = Bs_[o];
    const int mbase = r * 16 + ((lane >> 4) << 3);
#pragma unroll
    for (int t = 0; t < NT; ++t) {
      const int col = t * 16 + l16;
      const float bc = bias ? bias[col] : 0.0f;
#pragma unroll
      for (int i = 0; i < 8; ++i)
        C[(size_t)(mbase + i) * NC + col] = acc[t][i] + bc;
    }
  }
}

// ---------------------------------------------------------------------------
// Monotonic float<->uint encoding for atomicMax-based segment max.
// ---------------------------------------------------------------------------
__device__ __forceinline__ unsigned enc_f32(float f) {
  unsigned u = __float_as_uint(f);
  return (u & 0x80000000u) ? ~u : (u | 0x80000000u);
}
__device__ __forceinline__ float dec_f32(unsigned u) {
  return (u & 0x80000000u) ? __uint_as_float(u & 0x7FFFFFFFu)
                           : __uint_as_float(~u);
}

__global__ void zero_u32(unsigned* __restrict__ p, long long n) {
  long long t = (long long)blockIdx.x * blockDim.x + threadIdx.x;
  if (t < n) p[t] = 0u;
}

// alpha[e,h] = <q[dst], k[src]+e_edge> * scale ; segment max into nmax[dst,h]
__global__ void edge_alpha_max(const int* __restrict__ src, const int* __restrict__ dst,
                               const float* __restrict__ q, const float* __restrict__ kb,
                               const float* __restrict__ eb,
                               float* __restrict__ alpha, unsigned* __restrict__ nmax,
                               int E, int H, int Cc, int d, float scale) {
  int t = blockIdx.x * blockDim.x + threadIdx.x;
  if (t >= E * H) return;
  int e = t / H, h = t - e * H;
  int s = src[e], dn = dst[e];
  const float4* qp = (const float4*)(q  + (size_t)dn * d + h * Cc);
  const float4* kp = (const float4*)(kb + (size_t)s  * d + h * Cc);
  const float4* ep = (const float4*)(eb + (size_t)e  * d + h * Cc);
  float acc = 0.0f;
  for (int c = 0; c < Cc / 4; ++c) {
    float4 qv = qp[c], kv = kp[c], ev = ep[c];
    acc += qv.x * (kv.x + ev.x) + qv.y * (kv.y + ev.y) +
           qv.z * (kv.z + ev.z) + qv.w * (kv.w + ev.w);
  }
  acc *= scale;
  alpha[t] = acc;
  atomicMax(&nmax[(size_t)dn * H + h], enc_f32(acc));
}

// alpha[e,h] = exp(alpha - m[dst]); den[dst,h] += alpha
__global__ void edge_exp_den(const int* __restrict__ dst,
                             float* __restrict__ alpha,
                             const unsigned* __restrict__ nmax,
                             float* __restrict__ den, int E, int H) {
  int t = blockIdx.x * blockDim.x + threadIdx.x;
  if (t >= E * H) return;
  int e = t / H, h = t - e * H;
  int dn = dst[e];
  unsigned u = nmax[(size_t)dn * H + h];
  float m = (u == 0u) ? 0.0f : dec_f32(u);
  if (!(m == m) || m == __builtin_inff() || m == -__builtin_inff()) m = 0.0f;
  float ex = __expf(alpha[t] - m);
  alpha[t] = ex;
  atomicAdd(&den[(size_t)dn * H + h], ex);
}

// out[dst, h*C+c] += (v[src]+e)[h,c] * alpha/(den+eps)
__global__ void edge_scatter(const int* __restrict__ src, const int* __restrict__ dst,
                             const float* __restrict__ vb, const float* __restrict__ eb,
                             const float* __restrict__ alpha, const float* __restrict__ den,
                             float* __restrict__ out, int E, int H, int Cc, int d) {
  long long t = (long long)blockIdx.x * blockDim.x + threadIdx.x;
  long long total = (long long)E * H * Cc;
  if (t >= total) return;
  int c  = (int)(t % Cc);
  long long eh = t / Cc;
  int h  = (int)(eh % H);
  int e  = (int)(eh / H);
  int s = src[e], dn = dst[e];
  float a = alpha[(size_t)e * H + h] / (den[(size_t)dn * H + h] + 1e-16f);
  float msg = (vb[(size_t)s * d + h * Cc + c] + eb[(size_t)e * d + h * Cc + c]) * a;
  atomicAdd(&out[(size_t)dn * d + h * Cc + c], msg);
}

// Per-channel BatchNorm (batch statistics) + ReLU. One block per channel.
__global__ void bn_relu(const float* __restrict__ x,
                        const float* __restrict__ gamma, const float* __restrict__ beta,
                        float* __restrict__ y, int n, int d) {
  const int ch = blockIdx.x;
  __shared__ float ssum[256];
  __shared__ float ssq[256];
  float s = 0.0f, sq = 0.0f;
  for (int r = threadIdx.x; r < n; r += blockDim.x) {
    float v = x[(size_t)r * d + ch];
    s += v; sq += v * v;
  }
  ssum[threadIdx.x] = s; ssq[threadIdx.x] = sq;
  __syncthreads();
  for (int off = 128; off > 0; off >>= 1) {
    if (threadIdx.x < off) {
      ssum[threadIdx.x] += ssum[threadIdx.x + off];
      ssq[threadIdx.x]  += ssq[threadIdx.x + off];
    }
    __syncthreads();
  }
  const float mu  = ssum[0] / (float)n;
  const float var = ssq[0] / (float)n - mu * mu;
  const float rs  = rsqrtf(var + 1e-5f);
  const float g = gamma[ch], b = beta[ch];
  for (int r = threadIdx.x; r < n; r += blockDim.x) {
    float v = (x[(size_t)r * d + ch] - mu) * rs * g + b;
    y[(size_t)r * d + ch] = v > 0.0f ? v : 0.0f;
  }
}

__global__ void copy_f32(const float* __restrict__ src, float* __restrict__ dst, long long n) {
  long long t = (long long)blockIdx.x * blockDim.x + threadIdx.x;
  if (t < n) dst[t] = src[t];
}

// ---------------------------------------------------------------------------
// Host orchestration.
// d_in order (setup_inputs insertion order):
//  0:x 1:edge_index 2:edge_attr
//  3..11  L0 {Wq,Wk,Wv,We,Ws,bq,bk,bv,bs}
//  12..20 L1 {...}    21..29 L2 {...}
//  30,31 bn0 {gamma,beta}   32,33 bn1 {gamma,beta}
// ---------------------------------------------------------------------------
extern "C" void kernel_launch(void* const* d_in, const int* in_sizes, int n_in,
                              void* d_out, int out_size, void* d_ws, size_t ws_size,
                              hipStream_t stream) {
  const int IN = 64, HID = 256;
  const int N = in_sizes[0] / IN;       // 50000
  const int E = in_sizes[1] / 2;        // 800000

  const float* x         = (const float*)d_in[0];
  const int*   src       = (const int*)d_in[1];
  const int*   dst       = src + E;
  const float* edge_attr = (const float*)d_in[2];

  float* ws = (float*)d_ws;
  float* qB  = ws;                       // N*256
  float* kB  = qB + (size_t)N * HID;
  float* vB  = kB + (size_t)N * HID;
  float* h1  = vB + (size_t)N * HID;
  float* h2  = h1 + (size_t)N * HID;
  float* eB  = h2 + (size_t)N * HID;     // E*256
  float* alp = eB + (size_t)E * HID;     // E*4
  float* den = alp + (size_t)E * 4;      // N*4
  unsigned* nmax = (unsigned*)(den + (size_t)N * 4);

  const int nodeBlocks = (N / 16 + 7) / 8;   // tilesM=3125 -> 391 (clamped tail)
  const int edgeBlocks = (E / 16 + 7) / 8;   // tilesM=50000 -> 6250

  auto softmax_scatter = [&](int H, int Cc, int d, float* out) {
    long long nh = (long long)N * H;
    zero_u32<<<(unsigned)((nh + 255) / 256), 256, 0, stream>>>(nmax, nh);
    zero_u32<<<(unsigned)((nh + 255) / 256), 256, 0, stream>>>((unsigned*)den, nh);
    const float scale = rsqrtf((float)Cc);
    long long eh = (long long)E * H;
    edge_alpha_max<<<(unsigned)((eh + 255) / 256), 256, 0, stream>>>(
        src, dst, qB, kB, eB, alp, nmax, E, H, Cc, d, scale);
    edge_exp_den<<<(unsigned)((eh + 255) / 256), 256, 0, stream>>>(
        dst, alp, nmax, den, E, H);
    long long ehc = (long long)E * H * Cc;
    edge_scatter<<<(unsigned)((ehc + 255) / 256), 256, 0, stream>>>(
        src, dst, vB, eB, alp, den, out, E, H, Cc, d);
  };

  const float** P;
#define PF(i) ((const float*)P[i])

  // ---- Layer 0: cin=64, H=4, C=64 -> h1; BN+ReLU -> h2 -------------------
  P = (const float**)(d_in + 3);
  gemm_wmma<64, 256, 4><<<nodeBlocks, 256, 0, stream>>>(
      x, N, PF(0), PF(1), PF(2), PF(4), PF(5), PF(6), PF(7), PF(8),
      qB, kB, vB, h1);
  gemm_wmma<64, 256, 1><<<edgeBlocks, 256, 0, stream>>>(
      edge_attr, E, PF(3), nullptr, nullptr, nullptr,
      nullptr, nullptr, nullptr, nullptr, eB, nullptr, nullptr, nullptr);
  softmax_scatter(4, 64, 256, h1);
  bn_relu<<<HID, 256, 0, stream>>>(h1, (const float*)d_in[30], (const float*)d_in[31],
                                   h2, N, HID);

  // ---- Layer 1: cin=256, H=4, C=64 -> h1; BN+ReLU -> h2 ------------------
  P = (const float**)(d_in + 12);
  gemm_wmma<256, 256, 4><<<nodeBlocks, 256, 0, stream>>>(
      h2, N, PF(0), PF(1), PF(2), PF(4), PF(5), PF(6), PF(7), PF(8),
      qB, kB, vB, h1);
  gemm_wmma<64, 256, 1><<<edgeBlocks, 256, 0, stream>>>(
      edge_attr, E, PF(3), nullptr, nullptr, nullptr,
      nullptr, nullptr, nullptr, nullptr, eB, nullptr, nullptr, nullptr);
  softmax_scatter(4, 64, 256, h1);
  bn_relu<<<HID, 256, 0, stream>>>(h1, (const float*)d_in[32], (const float*)d_in[33],
                                   h2, N, HID);

  // ---- Layer 2: cin=256, H=1, C=32 -> d_out head [N,32] ------------------
  float* out_h = (float*)d_out;
  P = (const float**)(d_in + 21);
  gemm_wmma<256, 32, 4><<<nodeBlocks, 256, 0, stream>>>(
      h2, N, PF(0), PF(1), PF(2), PF(4), PF(5), PF(6), PF(7), PF(8),
      qB, kB, vB, out_h);
  gemm_wmma<64, 32, 1><<<edgeBlocks, 256, 0, stream>>>(
      edge_attr, E, PF(3), nullptr, nullptr, nullptr,
      nullptr, nullptr, nullptr, nullptr, eB, nullptr, nullptr, nullptr);
  softmax_scatter(1, 32, 32, out_h);

  // ---- Tuple tail: edge_attr passthrough ---------------------------------
  long long ecount = (long long)E * IN;
  copy_f32<<<(unsigned)((ecount + 255) / 256), 256, 0, stream>>>(
      edge_attr, out_h + (size_t)N * 32, ecount);
#undef PF
}
